// TransformerBlock_16758962389717
// MI455X (gfx1250) — matrix operand order repaired and checked by reading, not compiled, not run on hardware
//
#include <hip/hip_runtime.h>
#include <stdint.h>

// ---------------- constants ----------------
constexpr int BB   = 2;
constexpr int TT   = 2048;
constexpr int DD   = 1024;
constexpr int HH   = 16;
constexpr int QKVW = 1536;   // q(1024) | k(256) | v(256)
constexpr int FF   = 4096;

typedef __attribute__((ext_vector_type(16))) __bf16 bf16x16;
typedef __attribute__((ext_vector_type(8)))  float  f32x8;

union AFrag { unsigned u[8]; bf16x16 v; };

__device__ __forceinline__ unsigned short f2bf(float f) {
    unsigned u = __float_as_uint(f);
    unsigned r = (u + 0x7FFFu + ((u >> 16) & 1u)) >> 16;   // RNE
    return (unsigned short)r;
}

__device__ __forceinline__ f32x8 wmma_bf16(const AFrag& a, const AFrag& b, f32x8 c) {
    return __builtin_amdgcn_wmma_f32_16x16x32_bf16(false, a.v, false, b.v,
                                                   (short)0, c, false, false);
}

// ---- CDNA5 async global->LDS copy (ASYNCcnt-tracked) ----
typedef const __attribute__((address_space(3))) void* lds_cptr;
__device__ __forceinline__ unsigned lds_addr(const void* p) {
    return (unsigned)(uintptr_t)(lds_cptr)p;    // generic -> LDS offset
}
__device__ __forceinline__ void async_copy_b128(unsigned lds_off, const void* gaddr) {
    asm volatile("global_load_async_to_lds_b128 %0, %1, off"
                 :: "v"(lds_off), "v"(gaddr) : "memory");
}
__device__ __forceinline__ void wait_async0() {
    asm volatile("s_wait_asynccnt 0x0" ::: "memory");
}

// ---- wave32 XOR butterfly via ds_swizzle (no address VALU) ----
template <int MASK>
__device__ __forceinline__ float swz_xor(float x) {
    return __int_as_float(
        __builtin_amdgcn_ds_swizzle(__float_as_int(x), 0x1F | (MASK << 10)));
}
__device__ __forceinline__ float red_max16(float v) {
    v = fmaxf(v, swz_xor<1>(v)); v = fmaxf(v, swz_xor<2>(v));
    v = fmaxf(v, swz_xor<4>(v)); v = fmaxf(v, swz_xor<8>(v));
    return v;
}
__device__ __forceinline__ float red_sum16(float v) {
    v += swz_xor<1>(v); v += swz_xor<2>(v);
    v += swz_xor<4>(v); v += swz_xor<8>(v);
    return v;
}

// ---------------- LDS-tiled transpose + f32->bf16 ----------------
// src: K x N f32 (row-major) -> dst: N x K bf16 (row-major) = column-major weights
__global__ __launch_bounds__(256) void transpose_bf16_kernel(const float* __restrict__ src,
                                                             unsigned short* __restrict__ dst,
                                                             int K, int N) {
    __shared__ float tile[32][33];
    const int tx = threadIdx.x & 31, ty = threadIdx.x >> 5;   // 32 x 8
    const int bx = blockIdx.x * 32;   // n base
    const int by = blockIdx.y * 32;   // k base
#pragma unroll
    for (int j = 0; j < 4; j++)
        tile[ty + j * 8][tx] = src[(size_t)(by + ty + j * 8) * N + bx + tx];
    __syncthreads();
#pragma unroll
    for (int j = 0; j < 4; j++)
        dst[(size_t)(bx + ty + j * 8) * K + by + tx] = f2bf(tile[tx][ty + j * 8]);
}

__global__ __launch_bounds__(256) void pack_qkv_b_kernel(const float* bq, const float* bk,
                                                         const float* bv, float* dst) {
    int j = blockIdx.x * blockDim.x + threadIdx.x;
    if (j < QKVW) {
        if (j < DD)            dst[j] = bq[j];
        else if (j < DD + 256) dst[j] = bk[j - DD];
        else                   dst[j] = bv[j - DD - 256];
    }
}

// ---------------- layernorm -> bf16 ----------------
__global__ __launch_bounds__(256) void ln_kernel(const float* __restrict__ x,
                                                 const float* __restrict__ g,
                                                 const float* __restrict__ b,
                                                 unsigned short* __restrict__ out) {
    int row = blockIdx.x;
    int tid = threadIdx.x;
    const float* xr = x + (size_t)row * DD;
    float v[4];
    float s = 0.f, s2 = 0.f;
#pragma unroll
    for (int i = 0; i < 4; i++) {
        v[i] = xr[tid + i * 256];
        s += v[i]; s2 += v[i] * v[i];
    }
    __shared__ float rs[256], rs2[256];
    rs[tid] = s; rs2[tid] = s2;
    __syncthreads();
    for (int o = 128; o > 0; o >>= 1) {
        if (tid < o) { rs[tid] += rs[tid + o]; rs2[tid] += rs2[tid + o]; }
        __syncthreads();
    }
    float mean = rs[0] * (1.0f / DD);
    float var  = rs2[0] * (1.0f / DD) - mean * mean;
    float rstd = rsqrtf(var + 1e-5f);
#pragma unroll
    for (int i = 0; i < 4; i++) {
        int c = tid + i * 256;
        out[(size_t)row * DD + c] = f2bf(g[c] * (v[i] - mean) * rstd + b[c]);
    }
}

// ---------------- double-buffered async WMMA GEMM ----------------
// C[M,N] = A[M,K](bf16, row-major) * Wt[N,K](bf16, col-major weights) + bias
// EPI 0: store bf16    EPI 1: exact-erf GELU -> bf16    EPI 2: resid + acc -> f32
constexpr int AS_SZ = 128 * 40;   // A tile buffer, row stride 40 halves
constexpr int WS_SZ = 64 * 40;    // W tile buffer, row stride 40 halves

template <int EPI>
__global__ __launch_bounds__(256) void gemm_kernel(const unsigned short* __restrict__ A,
                                                   const unsigned short* __restrict__ Wt,
                                                   const float* __restrict__ bias,
                                                   const float* __restrict__ resid,
                                                   void* __restrict__ outp,
                                                   int M, int N, int K) {
    __shared__ unsigned short SM[2 * AS_SZ + 2 * WS_SZ];   // 30720 B
    const int tid  = threadIdx.x;
    const int lane = tid & 31, ln = lane & 15, hi = lane >> 4;
    const int wave = tid >> 5, wm = wave >> 1, wn = wave & 1;
    const int m0 = blockIdx.y * 128, n0 = blockIdx.x * 64;

    f32x8 acc[2][2] = {};

    // async staging assignments (verbatim row copies, 16B aligned)
    const int arow = tid >> 1, apart = tid & 1;    // A: 128 rows x 64B, 32B per thread
    const int wrow = tid >> 2, wpart = tid & 3;    // W: 64 rows x 64B, 16B per thread

    auto issue_tile = [&](int buf, int kt) {
        unsigned short* Asb = &SM[buf * AS_SZ];
        unsigned short* Wsb = &SM[2 * AS_SZ + buf * WS_SZ];
        const unsigned short* ga = A  + (size_t)(m0 + arow) * K + kt + apart * 16;
        const unsigned short* gw = Wt + (size_t)(n0 + wrow) * K + kt + wpart * 8;
        unsigned la = lds_addr(&Asb[arow * 40 + apart * 16]);
        async_copy_b128(la,      ga);
        async_copy_b128(la + 16, ga + 8);
        async_copy_b128(lds_addr(&Wsb[wrow * 40 + wpart * 8]), gw);
    };

    issue_tile(0, 0);
    int buf = 0;
    for (int kt = 0; kt < K; kt += 32) {
        wait_async0();          // my copies for `buf` have landed in LDS
        __syncthreads();        // everyone's copies landed; prev-buffer reads done
        if (kt + 32 < K) issue_tile(buf ^ 1, kt + 32);

        const unsigned short* Asb = &SM[buf * AS_SZ];
        const unsigned short* Wsb = &SM[2 * AS_SZ + buf * WS_SZ];
        AFrag a[2], bfr[2];
#pragma unroll
        for (int i = 0; i < 2; i++) {    // A fragments (16x32, contiguous k)
            int m = wm * 32 + i * 16 + ln;
#pragma unroll
            for (int j = 0; j < 8; j++) {
                int kk = hi * 16 + j * 2;
                a[i].u[j] = *(const unsigned*)&Asb[m * 40 + kk];
            }
        }
#pragma unroll
        for (int j = 0; j < 2; j++) {    // B fragments: lane = column, contiguous k
            int n = wn * 32 + j * 16 + ln;
#pragma unroll
            for (int p = 0; p < 8; p++)
                bfr[j].u[p] = *(const unsigned*)&Wsb[n * 40 + hi * 16 + p * 2];
        }
#pragma unroll
        for (int i = 0; i < 2; i++)
#pragma unroll
            for (int j = 0; j < 2; j++)
                acc[i][j] = wmma_bf16(a[i], bfr[j], acc[i][j]);
        buf ^= 1;
    }

#pragma unroll
    for (int i = 0; i < 2; i++)
#pragma unroll
        for (int j = 0; j < 2; j++)
#pragma unroll
            for (int r = 0; r < 8; r++) {
                int m = m0 + wm * 32 + i * 16 + r + hi * 8;
                int n = n0 + wn * 32 + j * 16 + ln;
                float v = acc[i][j][r] + bias[n];
                size_t idx = (size_t)m * N + n;
                if (EPI == 0) {
                    ((unsigned short*)outp)[idx] = f2bf(v);
                } else if (EPI == 1) {
                    float gv = 0.5f * v * (1.0f + erff(v * 0.70710678118654752f));
                    ((unsigned short*)outp)[idx] = f2bf(gv);
                } else {
                    ((float*)outp)[idx] = resid[idx] + v;
                }
            }
}

// ---------------- flash attention (GQA, causal) ----------------
// One block = one (batch, head, 128-row q tile). 8 waves x 16 q rows each.
__global__ __launch_bounds__(256) void attn_kernel(const unsigned short* __restrict__ qkv,
                                                   unsigned short* __restrict__ out) {
    __shared__ unsigned short Ks[32 * 72];        // K tile [key][d], stride 72
    __shared__ unsigned short Vt[64 * 40];        // V tile transposed [d][key], stride 40
    __shared__ unsigned short Ps[8 * 16 * 34];    // per-wave P buffer [16][32+2]

    const int tid  = threadIdx.x;
    const int wave = tid >> 5, lane = tid & 31, ln = lane & 15, hi = lane >> 4;
    const int bid = blockIdx.x;
    const int qt = bid & 15;             // T/128 = 16 tiles
    const int h  = (bid >> 4) & 15;
    const int b  = bid >> 8;
    const int g  = h >> 2;               // 4 heads per group
    const size_t rowOff = (size_t)b * TT;
    const int qrow0 = qt * 128 + wave * 16;
    const int kvoff = DD + g * 64;
    const int voff  = DD + 256 + g * 64;

    // Q A-fragments, loaded once (rows qrow0..+15, d=0..63)
    AFrag aq[2];
    {
        const unsigned short* qbase = qkv + (rowOff + qrow0 + ln) * QKVW + h * 64;
#pragma unroll
        for (int i = 0; i < 2; i++)
#pragma unroll
            for (int j = 0; j < 8; j++) {
                int kk = i * 32 + hi * 16 + j * 2;
                aq[i].u[j] = *(const unsigned*)(qbase + kk);
            }
    }

    float m_run[8], l_run[8];
#pragma unroll
    for (int r = 0; r < 8; r++) { m_run[r] = -1e30f; l_run[r] = 0.f; }
    f32x8 o[4] = {};

    unsigned short* Pw = &Ps[wave * 16 * 34];
    const int kend = qt * 128 + 128;

    for (int kk0 = 0; kk0 < kend; kk0 += 32) {
        __syncthreads();    // previous block's LDS reads complete
        {   // K tile (32 keys x 64 d): async global->LDS, 16B per thread
            int key = tid >> 3, part = tid & 7;
            async_copy_b128(lds_addr(&Ks[key * 72 + part * 8]),
                            qkv + (rowOff + kk0 + key) * QKVW + kvoff + part * 8);
        }
        {   // V tile transposed (manual: async can't transpose)
            int key = tid & 31, db = (tid >> 5) * 8;
            uint4 d = *(const uint4*)(qkv + (rowOff + kk0 + key) * QKVW + voff + db);
            const unsigned short* hp = (const unsigned short*)&d;
#pragma unroll
            for (int i = 0; i < 8; i++) Vt[(db + i) * 40 + key] = hp[i];
        }
        wait_async0();
        __syncthreads();

        if (kk0 <= qrow0 + 15) {    // wave-uniform: skip fully-masked key blocks
            // S = Q K^T  (two 16x16 score tiles, d=64 chained over 2 wmma)
            f32x8 s[2] = {};
#pragma unroll
            for (int t = 0; t < 2; t++) {
                AFrag bk0, bk1;
                int krow = t * 16 + ln;
#pragma unroll
                for (int p = 0; p < 8; p++) {
                    bk0.u[p] = *(const unsigned*)&Ks[krow * 72 + hi * 16 + p * 2];
                    bk1.u[p] = *(const unsigned*)&Ks[krow * 72 + 32 + hi * 16 + p * 2];
                }
                s[t] = wmma_bf16(aq[0], bk0, s[t]);
                s[t] = wmma_bf16(aq[1], bk1, s[t]);
            }
            // scale + causal mask, online softmax (ds_swizzle butterflies)
            float mt[8];
#pragma unroll
            for (int r = 0; r < 8; r++) {
                int row = qrow0 + r + hi * 8;
#pragma unroll
                for (int t = 0; t < 2; t++) {
                    float sv = s[t][r] * 0.125f;            // 1/sqrt(64)
                    int key = kk0 + t * 16 + ln;
                    if (key > row) sv = -1e30f;
                    s[t][r] = sv;
                }
                mt[r] = red_max16(fmaxf(s[0][r], s[1][r]));
            }
            float alpha[8];
#pragma unroll
            for (int r = 0; r < 8; r++) {
                float mn = fmaxf(m_run[r], mt[r]);
                alpha[r] = __expf(m_run[r] - mn);
                m_run[r] = mn;
            }
#pragma unroll
            for (int r = 0; r < 8; r++) {
                float p0 = __expf(s[0][r] - m_run[r]);
                float p1 = __expf(s[1][r] - m_run[r]);
                s[0][r] = p0; s[1][r] = p1;
                l_run[r] = l_run[r] * alpha[r] + red_sum16(p0 + p1);
            }
#pragma unroll
            for (int t2 = 0; t2 < 4; t2++)
#pragma unroll
                for (int r = 0; r < 8; r++) o[t2][r] *= alpha[r];

            // P (C-layout) -> LDS -> A-fragment layout
#pragma unroll
            for (int t = 0; t < 2; t++)
#pragma unroll
                for (int r = 0; r < 8; r++)
                    Pw[(r + hi * 8) * 34 + t * 16 + ln] = f2bf(s[t][r]);
            AFrag pa;
#pragma unroll
            for (int j = 0; j < 8; j++) {
                int kk = hi * 16 + j * 2;
                pa.u[j] = *(const unsigned*)&Pw[ln * 34 + kk];
            }
            // O += P * V
#pragma unroll
            for (int t2 = 0; t2 < 4; t2++) {
                AFrag bv;
#pragma unroll
                for (int p = 0; p < 8; p++)
                    bv.u[p] = *(const unsigned*)&Vt[(t2 * 16 + ln) * 40 + hi * 16 + p * 2];
                o[t2] = wmma_bf16(pa, bv, o[t2]);
            }
        }
    }

    // normalize + store bf16
#pragma unroll
    for (int r = 0; r < 8; r++) {
        float inv = 1.0f / l_run[r];
        size_t row = rowOff + qrow0 + r + hi * 8;
#pragma unroll
        for (int t2 = 0; t2 < 4; t2++)
            out[row * DD + h * 64 + t2 * 16 + ln] = f2bf(o[t2][r] * inv);
    }
}

// ---------------- launch ----------------
extern "C" void kernel_launch(void* const* d_in, const int* in_sizes, int n_in,
                              void* d_out, int out_size, void* d_ws, size_t ws_size,
                              hipStream_t stream) {
    const float* x   = (const float*)d_in[0];
    const float* g1  = (const float*)d_in[1];
    const float* bt1 = (const float*)d_in[2];
    const float* wq  = (const float*)d_in[3];
    const float* bq  = (const float*)d_in[4];
    const float* wk  = (const float*)d_in[5];
    const float* bk  = (const float*)d_in[6];
    const float* wv  = (const float*)d_in[7];
    const float* bv  = (const float*)d_in[8];
    const float* wo  = (const float*)d_in[9];
    const float* bo  = (const float*)d_in[10];
    const float* g2  = (const float*)d_in[11];
    const float* bt2 = (const float*)d_in[12];
    const float* w1  = (const float*)d_in[13];
    const float* b1  = (const float*)d_in[14];
    const float* w2  = (const float*)d_in[15];
    const float* b2  = (const float*)d_in[16];
    (void)in_sizes; (void)n_in; (void)out_size; (void)ws_size;

    char* ws = (char*)d_ws;
    size_t off = 0;
    auto carve = [&](size_t bytes) -> void* {
        void* p = ws + off;
        off = (off + bytes + 255) & ~(size_t)255;
        return p;
    };
    const size_t MT = (size_t)BB * TT;   // 4096 tokens
    unsigned short* h_bf   = (unsigned short*)carve(MT * DD * 2);
    unsigned short* wqkv_t = (unsigned short*)carve((size_t)QKVW * DD * 2);  // [n][k]
    float*          bqkv   = (float*)carve(QKVW * 4);
    unsigned short* wo_t   = (unsigned short*)carve((size_t)DD * DD * 2);    // [n][k]
    unsigned short* w1_t   = (unsigned short*)carve((size_t)FF * DD * 2);    // [n][k]
    unsigned short* w2_t   = (unsigned short*)carve((size_t)DD * FF * 2);    // [n][k]
    unsigned short* qkv    = (unsigned short*)carve(MT * QKVW * 2);
    unsigned short* attno  = (unsigned short*)carve(MT * DD * 2);
    float*          x1     = (float*)carve(MT * DD * 4);
    unsigned short* h2     = (unsigned short*)carve(MT * DD * 2);
    unsigned short* a1     = (unsigned short*)carve(MT * FF * 2);

    // weight transposition to bf16 col-major (one-time per launch, LDS-tiled)
    transpose_bf16_kernel<<<dim3(DD / 32, DD / 32), 256, 0, stream>>>(wq, wqkv_t, DD, DD);
    transpose_bf16_kernel<<<dim3(256 / 32, DD / 32), 256, 0, stream>>>(wk, wqkv_t + (size_t)DD * DD, DD, 256);
    transpose_bf16_kernel<<<dim3(256 / 32, DD / 32), 256, 0, stream>>>(wv, wqkv_t + (size_t)(DD + 256) * DD, DD, 256);
    transpose_bf16_kernel<<<dim3(DD / 32, DD / 32), 256, 0, stream>>>(wo, wo_t, DD, DD);
    transpose_bf16_kernel<<<dim3(FF / 32, DD / 32), 256, 0, stream>>>(w1, w1_t, DD, FF);
    transpose_bf16_kernel<<<dim3(DD / 32, FF / 32), 256, 0, stream>>>(w2, w2_t, FF, DD);
    pack_qkv_b_kernel<<<6, 256, 0, stream>>>(bq, bk, bv, bqkv);

    // LN1 -> h
    ln_kernel<<<(unsigned)MT, 256, 0, stream>>>(x, g1, bt1, h_bf);
    // fused QKV projection
    gemm_kernel<0><<<dim3(QKVW / 64, MT / 128), 256, 0, stream>>>(
        h_bf, wqkv_t, bqkv, nullptr, qkv, (int)MT, QKVW, DD);
    // causal GQA flash attention
    attn_kernel<<<BB * HH * (TT / 128), 256, 0, stream>>>(qkv, attno);
    // output projection + residual -> x1 (f32)
    gemm_kernel<2><<<dim3(DD / 64, MT / 128), 256, 0, stream>>>(
        attno, wo_t, bo, x, x1, (int)MT, DD, DD);
    // LN2 -> h2
    ln_kernel<<<(unsigned)MT, 256, 0, stream>>>(x1, g2, bt2, h2);
    // FFN1 + GELU
    gemm_kernel<1><<<dim3(FF / 64, MT / 128), 256, 0, stream>>>(
        h2, w1_t, b1, nullptr, a1, (int)MT, FF, DD);
    // FFN2 + residual -> d_out (f32)
    gemm_kernel<2><<<dim3(DD / 64, MT / 128), 256, 0, stream>>>(
        a1, w2_t, b2, x1, (float*)d_out, (int)MT, DD, FF);
}
